// GRU_70918499991697
// MI455X (gfx1250) — matrix operand order repaired and checked by reading, not compiled, not run on hardware
//
#include <hip/hip_runtime.h>

// ---------------------------------------------------------------------------
// GRU on MI455X (gfx1250): bf16 WMMA GEMMs, async global->LDS staging with
// double buffering, split-K recurrent GEMM, fused gate kernel.
// Reference shapes: B=128, I=768, T=128, H=768, G=3H=2304.
// ---------------------------------------------------------------------------

typedef __bf16 bf16;
typedef __attribute__((ext_vector_type(16))) bf16 v16bf;
typedef __attribute__((ext_vector_type(8)))  bf16 v8bf;
typedef __attribute__((ext_vector_type(8)))  float v8f;
typedef __attribute__((ext_vector_type(4)))  int   v4i;

#define GRU_B 128
#define GRU_I 768
#define GRU_T 128
#define GRU_H 768
#define GRU_G 2304   // 3*H

#define BM 128
#define BN 128
#define BK 32
#define SPLITK 4     // k-split for the recurrent (scan) GEMM

#if __has_builtin(__builtin_amdgcn_global_load_async_to_lds_b128)
#define GRU_HAS_ASYNC 1
#else
#define GRU_HAS_ASYNC 0
#endif

// 16B global -> LDS copy (async when available).
__device__ __forceinline__ void gru_copy16(const bf16* __restrict__ g,
                                           bf16* __restrict__ l) {
#if GRU_HAS_ASYNC
  __builtin_amdgcn_global_load_async_to_lds_b128(
      (v4i __attribute__((address_space(1)))*)g,
      (v4i __attribute__((address_space(3)))*)l, 0, 0);
#else
  *(v8bf*)l = *(const v8bf*)g;
#endif
}

template <int N>
__device__ __forceinline__ void gru_wait_async() {
#if GRU_HAS_ASYNC
#if __has_builtin(__builtin_amdgcn_s_wait_asynccnt)
  __builtin_amdgcn_s_wait_asynccnt(N);
#else
  asm volatile("s_wait_asynccnt %0" ::"i"(N) : "memory");
#endif
#endif
}

// ---------------------------------------------------------------------------
// Tiled GEMM: C[M,N] = A[M,K(range)] * W[N,K(range)]^T (+ bias[N] if given).
// A,W bf16 row-major; C fp32 row-major. blockIdx.z selects a K-split chunk;
// each split writes its own partial slab C + z*M*N.
// 256 threads = 8 waves (2x4 wave grid); wave owns 64x32 = 4x2 WMMA tiles.
// Double-buffered LDS filled with async global->LDS copies.
// ---------------------------------------------------------------------------
__global__ __launch_bounds__(256)
void gru_wmma_gemm(const bf16* __restrict__ A, const bf16* __restrict__ W,
                   const float* __restrict__ bias, float* __restrict__ C,
                   int M, int N, int K) {
  __shared__ __align__(16) bf16 lsA[2][BM * BK];
  __shared__ __align__(16) bf16 lsW[2][BN * BK];

  const int tid   = threadIdx.x;
  const int lane  = tid & 31;
  const int wave  = tid >> 5;
  const int waveM = wave >> 2;       // 0..1
  const int waveN = wave & 3;        // 0..3
  const int half  = lane >> 4;       // 0/1 (lane half)
  const int l16   = lane & 15;

  const int m0 = blockIdx.y * BM;
  const int n0 = blockIdx.x * BN;

  // K-split chunk handled by this block.
  const int klen  = K / gridDim.z;
  const int kbase = blockIdx.z * klen;
  const int iters = klen / BK;
  float* __restrict__ Cp = C + (size_t)blockIdx.z * M * N;

  // Staging: 1024 16B chunks per tile pair, 2 A + 2 W chunks per thread.
  const int c0row = tid >> 2;             // chunk row for s=0 (0..63)
  const int c0col = (tid & 3) << 3;       // chunk col (bf16 elems)

  auto stage = [&](int kt, int buf) {
    #pragma unroll
    for (int s = 0; s < 2; ++s) {
      const int row = c0row + s * 64;
      gru_copy16(&A[(size_t)(m0 + row) * K + kt + c0col],
                 &lsA[buf][row * BK + c0col]);
      gru_copy16(&W[(size_t)(n0 + row) * K + kt + c0col],
                 &lsW[buf][row * BK + c0col]);
    }
  };

  v8f acc[4][2] = {};

  stage(kbase, 0);                         // prologue fill, buffer 0

  for (int it = 0; it < iters; ++it) {
    const int buf = it & 1;
    if (it + 1 < iters) {
      __syncthreads();                     // buf^1 free (readers of it-1 done)
      stage(kbase + (it + 1) * BK, buf ^ 1);
      gru_wait_async<4>();                 // this wave's stage-it loads landed
    } else {
      gru_wait_async<0>();
    }
    __syncthreads();                       // all waves' stage-it data visible

    // A fragments: lane holds row m, K = 16*half + {0..15} (contiguous).
    v16bf afrag[4];
    #pragma unroll
    for (int i = 0; i < 4; ++i) {
      const bf16* base =
          &lsA[buf][(waveM * 64 + i * 16 + l16) * BK + half * 16];
      v8bf lo = *(const v8bf*)(base);
      v8bf hi = *(const v8bf*)(base + 8);
      afrag[i] = __builtin_shufflevector(lo, hi, 0, 1, 2, 3, 4, 5, 6, 7,
                                         8, 9, 10, 11, 12, 13, 14, 15);
    }
    // B fragments: lane holds col n, K = 16*half + {0..15} (contiguous).
    v16bf wfrag[2];
    #pragma unroll
    for (int j = 0; j < 2; ++j) {
      const bf16* base =
          &lsW[buf][(waveN * 32 + j * 16 + l16) * BK + half * 16];
      v8bf lo = *(const v8bf*)(base);
      v8bf hi = *(const v8bf*)(base + 8);
      wfrag[j] = __builtin_shufflevector(lo, hi, 0, 1, 2, 3, 4, 5, 6, 7,
                                         8, 9, 10, 11, 12, 13, 14, 15);
    }

    #pragma unroll
    for (int i = 0; i < 4; ++i)
      #pragma unroll
      for (int j = 0; j < 2; ++j)
        acc[i][j] = __builtin_amdgcn_wmma_f32_16x16x32_bf16(
            false, afrag[i], false, wfrag[j], (short)0, acc[i][j],
            false, false);
  }

  // Epilogue: C/D layout -> lane half selects m = r + 8*half.
  #pragma unroll
  for (int i = 0; i < 4; ++i) {
    #pragma unroll
    for (int j = 0; j < 2; ++j) {
      const int n   = n0 + waveN * 32 + j * 16 + l16;
      const float bv = bias ? bias[n] : 0.0f;
      #pragma unroll
      for (int r = 0; r < 8; ++r) {
        const int m = m0 + waveM * 64 + i * 16 + half * 8 + r;
        Cp[(size_t)m * N + n] = acc[i][j][r] + bv;
      }
    }
  }
}

// ---------------------------------------------------------------------------
// x [B,I,T] fp32  ->  xbf [T,B,I] bf16  (32x32 LDS transpose tiles per b)
// ---------------------------------------------------------------------------
__global__ __launch_bounds__(256)
void gru_transpose_x(const float* __restrict__ x, bf16* __restrict__ xbf) {
  __shared__ float tile[32][33];
  const int b  = blockIdx.z;
  const int i0 = blockIdx.y * 32;
  const int t0 = blockIdx.x * 32;
  const int tx = threadIdx.x & 31;   // along T (coalesced read)
  const int ty = threadIdx.x >> 5;   // 0..7

  const float* xp = x + ((size_t)b * GRU_I + i0) * GRU_T + t0;
  #pragma unroll
  for (int p = 0; p < 4; ++p) {
    const int ii = ty + p * 8;
    tile[ii][tx] = xp[(size_t)ii * GRU_T + tx];
  }
  __syncthreads();
  #pragma unroll
  for (int p = 0; p < 4; ++p) {
    const int tt = ty + p * 8;
    xbf[((size_t)(t0 + tt) * GRU_B + b) * GRU_I + i0 + tx] =
        (bf16)tile[tx][tt];
  }
}

// ---------------------------------------------------------------------------
// fp32 -> bf16 elementwise (weights)
// ---------------------------------------------------------------------------
__global__ void gru_f32_to_bf16(const float* __restrict__ src,
                                bf16* __restrict__ dst, int n) {
  const int i = blockIdx.x * blockDim.x + threadIdx.x;
  if (i < n) dst[i] = (bf16)src[i];
}

// ---------------------------------------------------------------------------
// h0 -> h (fp32 + bf16 copies in workspace)
// ---------------------------------------------------------------------------
__global__ void gru_init_h(const float* __restrict__ h0,
                           float* __restrict__ hf, bf16* __restrict__ hb,
                           int n) {
  const int i = blockIdx.x * blockDim.x + threadIdx.x;
  if (i < n) {
    const float v = h0[i];
    hf[i] = v;
    hb[i] = (bf16)v;
  }
}

// ---------------------------------------------------------------------------
// Gate fusion for one time step. gh comes as SPLITK fp32 partial slabs
// (no bias); we sum them and add b_hh here. In-place h update is safe
// (1:1 element map).
//   r = sig(gi_r + gh_r); z = sig(gi_z + gh_z); n = tanh(gi_n + r*gh_n)
//   h' = (1-z)*n + z*h
// ---------------------------------------------------------------------------
__device__ __forceinline__ float gru_sigmoid(float x) {
  return 1.0f / (1.0f + __expf(-x));
}

__global__ __launch_bounds__(256)
void gru_gate(const float* __restrict__ gi_t, const float* __restrict__ gh,
              const float* __restrict__ b_hh,
              float* __restrict__ h_f32, bf16* __restrict__ h_bf,
              float* __restrict__ out_t) {
  const int idx = blockIdx.x * blockDim.x + threadIdx.x;  // 0..B*H-1
  const int b = idx / GRU_H;
  const int j = idx - b * GRU_H;
  const float* gib = gi_t + (size_t)b * GRU_G;
  const float* ghb = gh + (size_t)b * GRU_G;

  float ghr = b_hh[j];
  float ghz = b_hh[GRU_H + j];
  float ghn = b_hh[2 * GRU_H + j];
  #pragma unroll
  for (int s = 0; s < SPLITK; ++s) {
    const float* p = ghb + (size_t)s * GRU_B * GRU_G;
    ghr += p[j];
    ghz += p[GRU_H + j];
    ghn += p[2 * GRU_H + j];
  }

  const float r = gru_sigmoid(gib[j] + ghr);
  const float z = gru_sigmoid(gib[GRU_H + j] + ghz);
  const float n = tanhf(gib[2 * GRU_H + j] + r * ghn);
  const float h = (1.0f - z) * n + z * h_f32[idx];

  h_f32[idx] = h;
  h_bf[idx]  = (bf16)h;
  out_t[idx] = h;
}

// ---------------------------------------------------------------------------
// Host-side orchestration
// ---------------------------------------------------------------------------
static inline size_t align256(size_t v) { return (v + 255) & ~(size_t)255; }

extern "C" void kernel_launch(void* const* d_in, const int* in_sizes, int n_in,
                              void* d_out, int out_size, void* d_ws, size_t ws_size,
                              hipStream_t stream) {
  (void)in_sizes; (void)n_in; (void)out_size; (void)ws_size;

  const float* x    = (const float*)d_in[0];  // [B,I,T]
  const float* h0   = (const float*)d_in[1];  // [B,H]
  const float* w_ih = (const float*)d_in[2];  // [3H,I]
  const float* w_hh = (const float*)d_in[3];  // [3H,H]
  const float* b_ih = (const float*)d_in[4];  // [3H]
  const float* b_hh = (const float*)d_in[5];  // [3H]
  float* out = (float*)d_out;                 // [T,B,H]

  // Workspace layout
  char* ws = (char*)d_ws;
  size_t off = 0;
  float* gi = (float*)(ws + off);                    // [T,B,3H] fp32
  off = align256(off + (size_t)GRU_T * GRU_B * GRU_G * sizeof(float));
  bf16* xbf = (bf16*)(ws + off);                     // [T,B,I] bf16
  off = align256(off + (size_t)GRU_T * GRU_B * GRU_I * sizeof(bf16));
  bf16* wih_bf = (bf16*)(ws + off);                  // [3H,I] bf16
  off = align256(off + (size_t)GRU_G * GRU_I * sizeof(bf16));
  bf16* whh_bf = (bf16*)(ws + off);                  // [3H,H] bf16
  off = align256(off + (size_t)GRU_G * GRU_H * sizeof(bf16));
  float* gh = (float*)(ws + off);                    // [SPLITK,B,3H] fp32
  off = align256(off + (size_t)SPLITK * GRU_B * GRU_G * sizeof(float));
  float* h_f32 = (float*)(ws + off);                 // [B,H] fp32
  off = align256(off + (size_t)GRU_B * GRU_H * sizeof(float));
  bf16* h_bf = (bf16*)(ws + off);                    // [B,H] bf16

  // 1) Weights -> bf16 (already [N,K] row-major for the Wᵀ operand)
  {
    const int n = GRU_G * GRU_I;
    gru_f32_to_bf16<<<(n + 255) / 256, 256, 0, stream>>>(w_ih, wih_bf, n);
    gru_f32_to_bf16<<<(n + 255) / 256, 256, 0, stream>>>(w_hh, whh_bf, n);
  }

  // 2) x [B,I,T] -> bf16 [T,B,I]
  {
    dim3 grid(GRU_T / 32, GRU_I / 32, GRU_B);
    gru_transpose_x<<<grid, 256, 0, stream>>>(x, xbf);
  }

  // 3) h init
  {
    const int n = GRU_B * GRU_H;
    gru_init_h<<<(n + 255) / 256, 256, 0, stream>>>(h0, h_f32, h_bf, n);
  }

  // 4) gi = xs @ w_ihᵀ + b_ih   (M=16384, N=2304, K=768, no k-split)
  {
    dim3 grid(GRU_G / BN, (GRU_T * GRU_B) / BM, 1);
    gru_wmma_gemm<<<grid, 256, 0, stream>>>(xbf, wih_bf, b_ih, gi,
                                            GRU_T * GRU_B, GRU_G, GRU_I);
  }

  // 5) Sequential scan over T: split-K GEMM into partials, fused gate.
  {
    dim3 ggrid(GRU_G / BN, GRU_B / BM, SPLITK);        // (18, 1, 4)
    const int ngate = GRU_B * GRU_H;                   // 98304
    for (int t = 0; t < GRU_T; ++t) {
      gru_wmma_gemm<<<ggrid, 256, 0, stream>>>(h_bf, whh_bf, nullptr, gh,
                                               GRU_B, GRU_G, GRU_H);
      gru_gate<<<ngate / 256, 256, 0, stream>>>(
          gi + (size_t)t * GRU_B * GRU_G, gh, b_hh, h_f32, h_bf,
          out + (size_t)t * GRU_B * GRU_H);
    }
  }
}
